// GraphTransformer_71322226917731
// MI455X (gfx1250) — matrix operand order repaired and checked by reading, not compiled, not run on hardware
//
#include <hip/hip_runtime.h>
#include <hip/hip_bf16.h>
#include <stdint.h>

typedef __attribute__((ext_vector_type(16))) __bf16 v16bf;
typedef __attribute__((ext_vector_type(8)))  float  v8f;

#define N_NODES 50000   // multiple of 16 -> no row guards needed in GEMM
#define N_EDGES 400000
#define F_DIM   128
#define N_HEADS 4
#define H_DIM   32
#define L_DIM   64

// ---------------------------------------------------------------- helpers
__device__ __forceinline__ __bf16 f2bf(float f) {
  // round-to-nearest-even f32 -> bf16 via integer path
  unsigned u = __float_as_uint(f);
  unsigned r = (u + 0x7FFFu + ((u >> 16) & 1u)) >> 16;
  unsigned short s = (unsigned short)r;
  return *reinterpret_cast<__bf16*>(&s);
}

__device__ __forceinline__ void atomicMaxFloat(float* addr, float val) {
  // monotone mapping trick: positives compare as int-max, negatives as uint-min
  if (val >= 0.0f) atomicMax((int*)addr, __float_as_int(val));
  else             atomicMin((unsigned int*)addr, __float_as_uint(val));
}

// ---------------------------------------------------------------- utility kernels
__global__ void fill_f32_kernel(float* __restrict__ p, float v, int n) {
  int i = blockIdx.x * blockDim.x + threadIdx.x;
  if (i < n) p[i] = v;
}

__global__ void f32_to_bf16_kernel(const float* __restrict__ in, __bf16* __restrict__ out, int n) {
  int i = blockIdx.x * blockDim.x + threadIdx.x;
  if (i < n) out[i] = f2bf(in[i]);
}

// ---------------------------------------------------------------- weight repack
// Convert W[K,Nd] (f32, row-major) into WMMA B-fragment-major bf16:
// frag index f = (ktile*ntn + tn)*32 + lane ; 16 contiguous bf16 per (f).
// Lane layout (16-bit B, this code's contiguous K order): lanes 0-15 hold K=k0..k0+15 (elem i -> K=k0+i),
// lanes 16-31 hold K=k0+16..k0+31; column = tn*16 + (lane & 15).
__global__ void pack_b_frag_kernel(const float* __restrict__ W,
                                   __bf16* __restrict__ Bp,
                                   int K, int Nd) {
  const int t = blockIdx.x * blockDim.x + threadIdx.x;
  const int ntn = Nd >> 4;
  const int nkt = K >> 5;
  const int total = nkt * ntn * 32;
  if (t >= total) return;
  const int lane = t & 31;
  const int tn   = (t >> 5) % ntn;
  const int kt   = (t >> 5) / ntn;
  const int kbB  = (lane < 16) ? 0 : 16;
  const int col  = tn * 16 + (lane & 15);
  __bf16* dstp = Bp + (size_t)t * 16;
#pragma unroll
  for (int i = 0; i < 16; ++i)
    dstp[i] = f2bf(W[(size_t)(kt * 32 + kbB + i) * Nd + col]);
}

// ---------------------------------------------------------------- WMMA GEMM
// D[M,Nd] = A[M,K](bf16 row-major) @ W(K x Nd, pre-packed fragments) + bias[Nd]
// One 16x32 output tile per wave32 (two accumulators share one A fragment).
// K compile-time -> fully unrolled -> 2*(K/32) v_wmma per wave.
// Requires: M % 16 == 0, Nd % 32 == 0, bias != nullptr, EXEC all-1s (wave-uniform exit only).
template<int K>
__global__ void wmma_gemm_bf16_kernel(const __bf16* __restrict__ A,
                                      const __bf16* __restrict__ Bp,
                                      const float*  __restrict__ bias,
                                      float* __restrict__ D,
                                      int M, int Nd) {
  const int lane = threadIdx.x & 31;
  const int wave = threadIdx.x >> 5;
  const int tile = blockIdx.x * (blockDim.x >> 5) + wave;
  const int ntn  = Nd >> 4;        // 16-wide column tiles
  const int ntn2 = Nd >> 5;        // 32-wide per-wave tiles
  const int ntm  = M >> 4;
  if (tile >= ntm * ntn2) return;  // wave-uniform
  const int tm  = tile / ntn2;
  const int tnp = tile - tm * ntn2;

  const int arow = tm * 16 + (lane & 15);
  const int kbA  = (lane < 16) ? 0 : 16;     // contiguous K per lane half (matches packed B)
  const __bf16* Arow = A + (size_t)arow * K;

  v8f acc0 = {};
  v8f acc1 = {};
  constexpr int NKT = K / 32;
#pragma unroll
  for (int kt = 0; kt < NKT; ++kt) {
    const int k0 = kt * 32;
    // A fragment: elems 0..15 -> K=k0+kbA+0..15
    union { float4 f4[2]; v16bf v; } ua;
    ua.f4[0] = *(const float4*)(Arow + k0 + kbA);
    ua.f4[1] = *(const float4*)(Arow + k0 + kbA + 8);

    // two pre-packed B fragments: contiguous 32B per lane
    const v16bf b0 = *(const v16bf*)(Bp + ((size_t)(kt * ntn + 2 * tnp + 0) * 32 + lane) * 16);
    const v16bf b1 = *(const v16bf*)(Bp + ((size_t)(kt * ntn + 2 * tnp + 1) * 32 + lane) * 16);

    acc0 = __builtin_amdgcn_wmma_f32_16x16x32_bf16(false, ua.v, false, b0,
                                                   (short)0, acc0, false, false);
    acc1 = __builtin_amdgcn_wmma_f32_16x16x32_bf16(false, ua.v, false, b1,
                                                   (short)0, acc1, false, false);
  }

  const int r0   = tm * 16 + ((lane < 16) ? 0 : 8);
  const int col0 = tnp * 32 + (lane & 15);
  const float bv0 = bias[col0];
  const float bv1 = bias[col0 + 16];
#pragma unroll
  for (int r = 0; r < 8; ++r) {
    float* drow = D + (size_t)(r0 + r) * Nd;
    drow[col0]      = acc0[r] + bv0;
    drow[col0 + 16] = acc1[r] + bv1;
  }
}

// ---------------------------------------------------------------- edge kernels
// Pass 1: raw scores + segment max.  One wave per edge.
template<int H, int C>
__global__ void edge_scores_kernel(const float* __restrict__ q,
                                   const float* __restrict__ k,
                                   const long long* __restrict__ src,
                                   const long long* __restrict__ dst,
                                   float* __restrict__ scoresE,   // [E,H]
                                   float* __restrict__ m,         // [N,H] init -inf
                                   int E) {
  const int lane = threadIdx.x & 31;
  const int e = (blockIdx.x * blockDim.x + threadIdx.x) >> 5;
  if (e >= E) return;
  const int s = (int)src[e];
  const int d = (int)dst[e];
  constexpr int HC = H * C;
  constexpr int J  = HC / 32;

  float acc[H];
#pragma unroll
  for (int h = 0; h < H; ++h) acc[h] = 0.0f;
#pragma unroll
  for (int j = 0; j < J; ++j) {
    const int idx = lane + 32 * j;
    const int h = (32 * j) / C;   // compile-time after unroll (C multiple of 32)
    acc[h] += q[(size_t)d * HC + idx] * k[(size_t)s * HC + idx];
  }
#pragma unroll
  for (int h = 0; h < H; ++h)
#pragma unroll
    for (int off = 16; off >= 1; off >>= 1)
      acc[h] += __shfl_xor(acc[h], off, 32);

  if (lane == 0) {
    const float scale = rsqrtf((float)C);
#pragma unroll
    for (int h = 0; h < H; ++h) {
      const float sv = acc[h] * scale;
      scoresE[(size_t)e * H + h] = sv;
      atomicMaxFloat(&m[(size_t)d * H + h], sv);
    }
  }
}

// Pass 2: e = exp(s - max), segment-sum denominators.
__global__ void edge_exp_kernel(float* __restrict__ scoresE,
                                const float* __restrict__ m,
                                float* __restrict__ denom,
                                const long long* __restrict__ dst,
                                int E, int H) {
  int i = blockIdx.x * blockDim.x + threadIdx.x;
  if (i >= E * H) return;
  const int e = i / H;
  const int h = i - e * H;
  const int d = (int)dst[e];
  const float ex = __expf(scoresE[i] - m[(size_t)d * H + h]);
  scoresE[i] = ex;
  atomicAdd(&denom[(size_t)d * H + h], ex);
}

// Pass 3: weighted message scatter. One wave per edge, coalesced over features.
template<int H, int C>
__global__ void edge_scatter_kernel(const float* __restrict__ v,
                                    const long long* __restrict__ src,
                                    const long long* __restrict__ dst,
                                    const float* __restrict__ scoresE,
                                    const float* __restrict__ denom,
                                    float* __restrict__ out,       // [N,H*C] init 0
                                    int E) {
  const int lane = threadIdx.x & 31;
  const int e = (blockIdx.x * blockDim.x + threadIdx.x) >> 5;
  if (e >= E) return;
  const int s = (int)src[e];
  const int d = (int)dst[e];
  constexpr int HC = H * C;
  constexpr int J  = HC / 32;

  float w[H];
#pragma unroll
  for (int h = 0; h < H; ++h)
    w[h] = scoresE[(size_t)e * H + h] / denom[(size_t)d * H + h];

#pragma unroll
  for (int j = 0; j < J; ++j) {
    const int idx = lane + 32 * j;
    const int h = (32 * j) / C;
    atomicAdd(&out[(size_t)d * HC + idx], w[h] * v[(size_t)s * HC + idx]);
  }
}

// ---------------------------------------------------------------- epilogues
__global__ void finalize_elu_kernel(const float* __restrict__ attn,
                                    const float* __restrict__ skip,
                                    __bf16* __restrict__ hout, int n) {
  int i = blockIdx.x * blockDim.x + threadIdx.x;
  if (i >= n) return;
  float t = attn[i] + skip[i];
  t = (t > 0.0f) ? t : expm1f(t);
  hout[i] = f2bf(t);
}

__global__ void finalize_sigmoid_kernel(const float* __restrict__ attn,
                                        const float* __restrict__ skip,
                                        float* __restrict__ out, int n) {
  int i = blockIdx.x * blockDim.x + threadIdx.x;
  if (i >= n) return;
  const float t = attn[i] + skip[i];
  out[i] = 1.0f / (1.0f + __expf(-t));
}

// ---------------------------------------------------------------- launch
extern "C" void kernel_launch(void* const* d_in, const int* in_sizes, int n_in,
                              void* d_out, int out_size, void* d_ws, size_t ws_size,
                              hipStream_t stream) {
  const float*     x        = (const float*)d_in[0];
  const long long* ei       = (const long long*)d_in[1];   // int64 per reference
  const float*     Wqkv     = (const float*)d_in[2];
  const float*     bqkv     = (const float*)d_in[3];
  const float*     Wsk      = (const float*)d_in[4];
  const float*     bsk      = (const float*)d_in[5];
  const float*     Wqkv_end = (const float*)d_in[6];
  const float*     bqkv_end = (const float*)d_in[7];
  const float*     Wsk_end  = (const float*)d_in[8];
  const float*     bsk_end  = (const float*)d_in[9];
  (void)in_sizes; (void)n_in; (void)out_size; (void)ws_size;

  const long long* src = ei;
  const long long* dst = ei + N_EDGES;

  // bump allocator on workspace
  char* wp = (char*)d_ws;
  auto alloc = [&](size_t bytes) -> void* {
    void* r = (void*)wp;
    wp += (bytes + 255) & ~(size_t)255;
    return r;
  };
  __bf16* h_bf     = (__bf16*)alloc((size_t)N_NODES * F_DIM * sizeof(__bf16));
  float*  qb       = (float*) alloc((size_t)N_NODES * F_DIM * sizeof(float));
  float*  kb       = (float*) alloc((size_t)N_NODES * F_DIM * sizeof(float));
  float*  vb       = (float*) alloc((size_t)N_NODES * F_DIM * sizeof(float));
  float*  skip     = (float*) alloc((size_t)N_NODES * F_DIM * sizeof(float));
  float*  attn     = (float*) alloc((size_t)N_NODES * F_DIM * sizeof(float));
  float*  scoresE  = (float*) alloc((size_t)N_EDGES * N_HEADS * sizeof(float));
  float*  mbuf     = (float*) alloc((size_t)N_NODES * N_HEADS * sizeof(float));
  float*  denom    = (float*) alloc((size_t)N_NODES * N_HEADS * sizeof(float));
  __bf16* wqkv_bp  = (__bf16*)alloc((size_t)3 * 3 * F_DIM * F_DIM * sizeof(__bf16));
  __bf16* wsk_bp   = (__bf16*)alloc((size_t)3 * F_DIM * F_DIM * sizeof(__bf16));
  __bf16* wqe_bp   = (__bf16*)alloc((size_t)3 * F_DIM * L_DIM * sizeof(__bf16));
  __bf16* wse_bp   = (__bf16*)alloc((size_t)F_DIM * L_DIM * sizeof(__bf16));

  auto cdiv = [](int a, int b) { return (a + b - 1) / b; };
  const int TPB = 256;

  auto fill = [&](float* p, float v, int n) {
    fill_f32_kernel<<<cdiv(n, TPB), TPB, 0, stream>>>(p, v, n);
  };
  auto packW = [&](const float* W, __bf16* Bp, int Nd) {
    const int total = (F_DIM >> 5) * (Nd >> 4) * 32;
    pack_b_frag_kernel<<<cdiv(total, TPB), TPB, 0, stream>>>(W, Bp, F_DIM, Nd);
  };
  auto gemm = [&](const __bf16* A, const __bf16* Bp, const float* bias, float* D, int Nd) {
    const int tiles = (N_NODES >> 4) * (Nd >> 5);
    wmma_gemm_bf16_kernel<F_DIM><<<cdiv(tiles, TPB / 32), TPB, 0, stream>>>(
        A, Bp, bias, D, N_NODES, Nd);
  };

  // one-time conversions / weight repacks
  f32_to_bf16_kernel<<<cdiv(N_NODES * F_DIM, TPB), TPB, 0, stream>>>(
      x, h_bf, N_NODES * F_DIM);
  for (int w = 0; w < 9; ++w)
    packW(Wqkv + (size_t)w * F_DIM * F_DIM, wqkv_bp + (size_t)w * F_DIM * F_DIM, F_DIM);
  for (int l = 0; l < 3; ++l)
    packW(Wsk + (size_t)l * F_DIM * F_DIM, wsk_bp + (size_t)l * F_DIM * F_DIM, F_DIM);
  for (int w = 0; w < 3; ++w)
    packW(Wqkv_end + (size_t)w * F_DIM * L_DIM, wqe_bp + (size_t)w * F_DIM * L_DIM, L_DIM);
  packW(Wsk_end, wse_bp, L_DIM);

  const int warpBlocks = cdiv(N_EDGES * 32, TPB);

  // ---- 3 TransformerConv(H=4, C=32) layers + ELU ----
  for (int l = 0; l < 3; ++l) {
    fill(mbuf, -INFINITY, N_NODES * N_HEADS);
    fill(denom, 0.0f, N_NODES * N_HEADS);
    fill(attn, 0.0f, N_NODES * F_DIM);

    const size_t woff = (size_t)l * 3 * F_DIM * F_DIM;
    const size_t boff = (size_t)l * 3 * F_DIM;
    gemm(h_bf, wqkv_bp + woff + 0 * F_DIM * F_DIM, bqkv + boff + 0 * F_DIM, qb, F_DIM);
    gemm(h_bf, wqkv_bp + woff + 1 * F_DIM * F_DIM, bqkv + boff + 1 * F_DIM, kb, F_DIM);
    gemm(h_bf, wqkv_bp + woff + 2 * F_DIM * F_DIM, bqkv + boff + 2 * F_DIM, vb, F_DIM);
    gemm(h_bf, wsk_bp + (size_t)l * F_DIM * F_DIM, bsk + (size_t)l * F_DIM, skip, F_DIM);

    edge_scores_kernel<N_HEADS, H_DIM><<<warpBlocks, TPB, 0, stream>>>(
        qb, kb, src, dst, scoresE, mbuf, N_EDGES);
    edge_exp_kernel<<<cdiv(N_EDGES * N_HEADS, TPB), TPB, 0, stream>>>(
        scoresE, mbuf, denom, dst, N_EDGES, N_HEADS);
    edge_scatter_kernel<N_HEADS, H_DIM><<<warpBlocks, TPB, 0, stream>>>(
        vb, src, dst, scoresE, denom, attn, N_EDGES);

    finalize_elu_kernel<<<cdiv(N_NODES * F_DIM, TPB), TPB, 0, stream>>>(
        attn, skip, h_bf, N_NODES * F_DIM);
  }

  // ---- final TransformerConv(H=1, C=64) + sigmoid ----
  fill(mbuf, -INFINITY, N_NODES);
  fill(denom, 0.0f, N_NODES);
  fill(attn, 0.0f, N_NODES * L_DIM);

  gemm(h_bf, wqe_bp + 0 * F_DIM * L_DIM, bqkv_end + 0 * L_DIM, qb, L_DIM);
  gemm(h_bf, wqe_bp + 1 * F_DIM * L_DIM, bqkv_end + 1 * L_DIM, kb, L_DIM);
  gemm(h_bf, wqe_bp + 2 * F_DIM * L_DIM, bqkv_end + 2 * L_DIM, vb, L_DIM);
  gemm(h_bf, wse_bp, bsk_end, skip, L_DIM);

  edge_scores_kernel<1, L_DIM><<<warpBlocks, TPB, 0, stream>>>(
      qb, kb, src, dst, scoresE, mbuf, N_EDGES);
  edge_exp_kernel<<<cdiv(N_EDGES, TPB), TPB, 0, stream>>>(
      scoresE, mbuf, denom, dst, N_EDGES, 1);
  edge_scatter_kernel<1, L_DIM><<<warpBlocks, TPB, 0, stream>>>(
      vb, src, dst, scoresE, denom, attn, N_EDGES);

  finalize_sigmoid_kernel<<<cdiv(N_NODES * L_DIM, TPB), TPB, 0, stream>>>(
      attn, skip, (float*)d_out, N_NODES * L_DIM);
}
